// Attention_11501922419237
// MI455X (gfx1250) — matrix operand order repaired and checked by reading, not compiled, not run on hardware
//
#include <hip/hip_runtime.h>
#include <hip/hip_bf16.h>

// ---------- types ----------
typedef __attribute__((ext_vector_type(16))) __bf16 v16bf;
typedef __attribute__((ext_vector_type(8)))  float  v8f;

union FragU {
  v16bf v;
  uint4 q[2];
};

__device__ __forceinline__ unsigned short f2bf(float f) {
  unsigned int u = __float_as_uint(f);
  unsigned int r = (u + 0x7FFFu + ((u >> 16) & 1u)) >> 16; // RNE
  return (unsigned short)r;
}

// ---------- fp32 -> bf16 cast ----------
__global__ void cvt_f32_to_bf16(const float* __restrict__ in,
                                unsigned short* __restrict__ out, int n) {
  int i = blockIdx.x * blockDim.x + threadIdx.x;
  if (i < n) out[i] = f2bf(in[i]);
}

// ---------- LDS-staged, double-buffered WMMA GEMM core ----------
// Block tile 128(M) x 256(N), K-slab 64. 8 waves: 2(M) x 4(N), wave tile 64x64.
// A: MxK row-major (lda), Bt: NxK row-major (ldb).
// LDS per buffer: A tile 128 rows x 144B (128B data + 16B pad), B tile 256 rows x 144B.
#define MT 4
#define NT 4
#define KSLAB 64
#define ROWB 144                        // padded row stride in bytes
#define ATILEB (128 * ROWB)             // 18432 B
#define BTILEB (256 * ROWB)             // 36864 B
#define BUFB (ATILEB + BTILEB)          // 55296 B per buffer
#define SMEMB (2 * BUFB)                // 110592 B double buffered
#define PF_OPS 12                       // async ops per wave per slab (4 A + 8 B)

// Async DMA of a ROWS x 64 bf16 tile (row0.., k0..k0+63) into LDS.
// ROWS*8 16-byte chunks, distributed over 256 threads.
template <int ROWS>
__device__ __forceinline__ void async_copy_tile(
    const unsigned short* __restrict__ g, int ld, int row0, int k0,
    char* ldsbase, int tid) {
#pragma unroll
  for (int r = 0; r < ROWS / 32; ++r) {
    const int c = tid + 256 * r;       // chunk id
    const int row = c >> 3;            // 8 chunks per row
    const int col8 = c & 7;            // 8 bf16 per chunk
    const unsigned short* gp = g + (size_t)(row0 + row) * ld + k0 + col8 * 8;
    const unsigned lds = (unsigned)(uintptr_t)(ldsbase + row * ROWB + col8 * 16);
    asm volatile("global_load_async_to_lds_b128 %0, %1, off"
                 :: "v"(lds), "v"(gp) : "memory");
  }
}

__device__ __forceinline__ void wmma_block_tile_lds(
    const unsigned short* __restrict__ A, const unsigned short* __restrict__ Bt,
    int lda, int ldb, int Kdim, int m0blk, int n0blk, v8f acc[MT][NT], char* sm) {
  const int tid = (int)threadIdx.x;
  const int wave = tid >> 5;
  const int lane = tid & 31;
  const int hf = lane >> 4;
  const int mr = lane & 15;
  const int mw = (wave >> 2) * 64;     // wave M offset within block tile
  const int nw = (wave & 3) * 64;      // wave N offset within block tile
  const int nk = Kdim / KSLAB;

  // prefetch slab 0 into buffer 0
  async_copy_tile<128>(A, lda, m0blk, 0, sm, tid);
  async_copy_tile<256>(Bt, ldb, n0blk, 0, sm + ATILEB, tid);

  for (int kt = 0; kt < nk; ++kt) {
    char* cur = sm + (kt & 1) * BUFB;
    if (kt + 1 < nk) {
      char* nxt = sm + ((kt + 1) & 1) * BUFB;
      async_copy_tile<128>(A, lda, m0blk, (kt + 1) * KSLAB, nxt, tid);
      async_copy_tile<256>(Bt, ldb, n0blk, (kt + 1) * KSLAB, nxt + ATILEB, tid);
      asm volatile("s_wait_asynccnt 12" ::: "memory");  // slab kt landed
    } else {
      asm volatile("s_wait_asynccnt 0" ::: "memory");
    }
    __syncthreads();   // all waves' DMA for 'cur' visible

#pragma unroll
    for (int kk = 0; kk < 2; ++kk) {   // two K=32 WMMA steps per slab
      const int k0l = kk * 32;
      FragU a[MT], b[NT];
#pragma unroll
      for (int i = 0; i < MT; ++i) {
        // A 16x32 layout: lanes 0-15 K=0..15, lanes 16-31 K=16..31
        const uint4* p = reinterpret_cast<const uint4*>(
            cur + (mw + 16 * i + mr) * ROWB + (k0l + hf * 16) * 2);
        a[i].q[0] = p[0];
        a[i].q[1] = p[1];
      }
#pragma unroll
      for (int j = 0; j < NT; ++j) {
        // B 32x16 layout: lane n -> column n&15; lanes 0-15 K=0..15, lanes 16-31 K=16..31
        const uint4* p = reinterpret_cast<const uint4*>(
            cur + ATILEB + (nw + 16 * j + mr) * ROWB + (k0l + hf * 16) * 2);
        b[j].q[0] = p[0];
        b[j].q[1] = p[1];
      }
#pragma unroll
      for (int i = 0; i < MT; ++i)
#pragma unroll
        for (int j = 0; j < NT; ++j)
          acc[i][j] = __builtin_amdgcn_wmma_f32_16x16x32_bf16(
              false, a[i].v, false, b[j].v, (short)0, acc[i][j], false, false);
    }
    __syncthreads();   // done reading 'cur'; safe to DMA over it next round
  }
}

// ---------- QKV GEMM: qkv = X @ W^T + b; split-store Q, K row-major and V transposed ----------
__global__ __launch_bounds__(256) void qkv_gemm_kernel(
    const unsigned short* __restrict__ X, const unsigned short* __restrict__ Wt,
    const float* __restrict__ bias,
    unsigned short* __restrict__ Q, unsigned short* __restrict__ Kc,
    unsigned short* __restrict__ Vt) {
  extern __shared__ char smem[];
  const int wave = (int)(threadIdx.x >> 5);
  const int m0w = blockIdx.y * 128 + (wave >> 2) * 64;
  const int n0w = blockIdx.x * 256 + (wave & 3) * 64;

  v8f acc[MT][NT];
#pragma unroll
  for (int i = 0; i < MT; ++i)
#pragma unroll
    for (int j = 0; j < NT; ++j)
      acc[i][j] = (v8f){0.f, 0.f, 0.f, 0.f, 0.f, 0.f, 0.f, 0.f};

  wmma_block_tile_lds(X, Wt, 1024, 1024, 1024,
                      blockIdx.y * 128, blockIdx.x * 256, acc, smem);

  const int lane = threadIdx.x & 31;
  const int hf = lane >> 4;
  const int mr = lane & 15;

#pragma unroll
  for (int i = 0; i < MT; ++i) {
#pragma unroll
    for (int j = 0; j < NT; ++j) {
      const int e = n0w + 16 * j + mr;     // feature column in [0,3072)
      const float bv = bias[e];
#pragma unroll
      for (int v = 0; v < 8; ++v) {
        const int row = m0w + 16 * i + hf * 8 + v;   // token row in [0,8192)
        const unsigned short h = f2bf(acc[i][j][v] + bv);
        if (e < 1024) {
          Q[(size_t)row * 1024 + e] = h;
        } else if (e < 2048) {
          Kc[(size_t)row * 1024 + (e - 1024)] = h;
        } else {
          const int bt = row >> 11;
          const int n = row & 2047;
          Vt[((size_t)bt * 1024 + (e - 2048)) * 2048 + n] = h;  // d-major
        }
      }
    }
  }
}

// ---------- generic WMMA GEMM with f32 store: C = A @ Bt^T ----------
__global__ __launch_bounds__(256) void gemm_f32_kernel(
    const unsigned short* __restrict__ A, const unsigned short* __restrict__ Bt,
    float* __restrict__ C, int lda, int ldb, int ldc, int Kdim) {
  extern __shared__ char smem[];
  const int wave = (int)(threadIdx.x >> 5);
  const int m0w = blockIdx.y * 128 + (wave >> 2) * 64;
  const int n0w = blockIdx.x * 256 + (wave & 3) * 64;

  v8f acc[MT][NT];
#pragma unroll
  for (int i = 0; i < MT; ++i)
#pragma unroll
    for (int j = 0; j < NT; ++j)
      acc[i][j] = (v8f){0.f, 0.f, 0.f, 0.f, 0.f, 0.f, 0.f, 0.f};

  wmma_block_tile_lds(A, Bt, lda, ldb, Kdim,
                      blockIdx.y * 128, blockIdx.x * 256, acc, smem);

  const int lane = threadIdx.x & 31;
  const int hf = lane >> 4;
  const int mr = lane & 15;

#pragma unroll
  for (int i = 0; i < MT; ++i) {
#pragma unroll
    for (int j = 0; j < NT; ++j) {
      const int col = n0w + 16 * j + mr;
#pragma unroll
      for (int v = 0; v < 8; ++v) {
        const int row = m0w + 16 * i + hf * 8 + v;
        C[(size_t)row * ldc + col] = acc[i][j][v];
      }
    }
  }
}

// ---------- row softmax: S (2048x2048 f32) -> P (2048x2048 bf16) ----------
__global__ __launch_bounds__(256) void softmax_rows_kernel(
    const float* __restrict__ S, unsigned short* __restrict__ P) {
  const int row = blockIdx.x;
  const int tid = threadIdx.x;
  const float* srow = S + (size_t)row * 2048;

  float v[8];
  float m = -3.402823466e+38f;
#pragma unroll
  for (int j = 0; j < 8; ++j) {
    v[j] = srow[tid + 256 * j];
    m = fmaxf(m, v[j]);
  }
#pragma unroll
  for (int off = 16; off >= 1; off >>= 1)
    m = fmaxf(m, __shfl_xor(m, off, 32));

  __shared__ float smax[8];
  __shared__ float ssum[8];
  const int wid = tid >> 5;
  const int ln = tid & 31;
  if (ln == 0) smax[wid] = m;
  __syncthreads();
  float bm = smax[0];
#pragma unroll
  for (int w = 1; w < 8; ++w) bm = fmaxf(bm, smax[w]);

  float s = 0.f;
#pragma unroll
  for (int j = 0; j < 8; ++j) {
    v[j] = __expf(v[j] - bm);
    s += v[j];
  }
#pragma unroll
  for (int off = 16; off >= 1; off >>= 1)
    s += __shfl_xor(s, off, 32);
  if (ln == 0) ssum[wid] = s;
  __syncthreads();
  float bs = 0.f;
#pragma unroll
  for (int w = 0; w < 8; ++w) bs += ssum[w];
  const float inv = 1.0f / bs;

  unsigned short* prow = P + (size_t)row * 2048;
#pragma unroll
  for (int j = 0; j < 8; ++j)
    prow[tid + 256 * j] = f2bf(v[j] * inv);
}

// ---------- launch ----------
extern "C" void kernel_launch(void* const* d_in, const int* in_sizes, int n_in,
                              void* d_out, int out_size, void* d_ws, size_t ws_size,
                              hipStream_t stream) {
  (void)in_sizes; (void)n_in; (void)out_size; (void)ws_size;
  const float* x    = (const float*)d_in[0];  // (4,2048,1024)
  const float* W    = (const float*)d_in[1];  // (3072,1024)
  const float* bias = (const float*)d_in[2];  // (3072,)
  float* out = (float*)d_out;                 // (4,2048,1024)

  char* ws = (char*)d_ws;
  unsigned short* x16 = (unsigned short*)ws;                          // 8192*1024 bf16
  unsigned short* W16 = (unsigned short*)(ws + 16777216);             // 3072*1024 bf16
  unsigned short* Q   = (unsigned short*)(ws + 16777216 + 6291456);   // 8192*1024 bf16
  unsigned short* Kc  = Q + (size_t)8192 * 1024;                      // 8192*1024 bf16
  unsigned short* Vt  = Kc + (size_t)8192 * 1024;                     // 4*1024*2048 bf16 (d-major)
  float* S            = (float*)(Vt + (size_t)8192 * 1024);           // 2048*2048 f32 (reused/batch)
  unsigned short* P   = (unsigned short*)(S + (size_t)2048 * 2048);   // 2048*2048 bf16 (reused)

  const int nX = 8192 * 1024;
  const int nW = 3072 * 1024;
  cvt_f32_to_bf16<<<(nX + 255) / 256, 256, 0, stream>>>(x, x16, nX);
  cvt_f32_to_bf16<<<(nW + 255) / 256, 256, 0, stream>>>(W, W16, nW);

  // QKV projection: M=8192, N=3072, K=1024
  qkv_gemm_kernel<<<dim3(3072 / 256, 8192 / 128), 256, SMEMB, stream>>>(
      x16, W16, bias, Q, Kc, Vt);

  for (int b = 0; b < 4; ++b) {
    const unsigned short* Qb  = Q  + (size_t)b * 2048 * 1024;
    const unsigned short* Kb  = Kc + (size_t)b * 2048 * 1024;
    const unsigned short* Vtb = Vt + (size_t)b * 1024 * 2048;
    float* outb = out + (size_t)b * 2048 * 1024;

    // S = Qb @ Kb^T : M=2048, N=2048, K=1024
    gemm_f32_kernel<<<dim3(2048 / 256, 2048 / 128), 256, SMEMB, stream>>>(
        Qb, Kb, S, 1024, 1024, 2048, 1024);

    // P = softmax(S) rows
    softmax_rows_kernel<<<2048, 256, 0, stream>>>(S, P);

    // out = P @ V : M=2048, N=1024, K=2048 ; B fragments from Vt (d-major)
    gemm_f32_kernel<<<dim3(1024 / 256, 2048 / 128), 256, SMEMB, stream>>>(
        P, Vtb, outb, 2048, 2048, 1024, 2048);
  }
}
